// SelfAttention_40931038331121
// MI455X (gfx1250) — matrix operand order repaired and checked by reading, not compiled, not run on hardware
//
#include <hip/hip_runtime.h>

// ---------------------------------------------------------------------------
// MI455X / gfx1250 dilated sliding-window self-attention.
// Math path: bf16 -> V_WMMA_F32_16X16X32_BF16 (f32 accumulate).
// Data path: double-buffered GLOBAL_LOAD_ASYNC_TO_LDS_B128 (ASYNCcnt) staging.
// ---------------------------------------------------------------------------

typedef __attribute__((ext_vector_type(16))) __bf16 v16bf;
typedef __attribute__((ext_vector_type(8)))  float  v8f;

union FragB { v16bf v; unsigned int u[8]; uint4 q[2]; };
union FragC { v8f  v; float f[8]; };

__device__ __forceinline__ v8f wmma_bf16(v16bf a, v16bf b, v8f c) {
  // 8 args: (neg_a, A, neg_b, B, c_mod, C, reuse_a, reuse_b)
  return __builtin_amdgcn_wmma_f32_16x16x32_bf16(false, a, false, b, (short)0, c,
                                                 false, false);
}

__device__ __forceinline__ unsigned short f2bf(float x) {
  unsigned int u = __float_as_uint(x);
  u += 0x7fffu + ((u >> 16) & 1u);          // round-to-nearest-even
  return (unsigned short)(u >> 16);
}

// Low 32 bits of a generic shared pointer are the LDS byte offset
// (flat LDS aperture keeps the offset in addr[31:0]).
__device__ __forceinline__ unsigned lds_addr32(const void* p) {
  return (unsigned)(unsigned long long)p;
}

// Async global -> LDS DMA, 16B per lane, GVS addressing (SGPR base + VGPR off).
__device__ __forceinline__ void async_load_b128(unsigned lds_byte,
                                                unsigned gl_byte_off,
                                                const void* sbase) {
  asm volatile("global_load_async_to_lds_b128 %0, %1, %2"
               :: "v"(lds_byte), "v"(gl_byte_off),
                  "s"((unsigned long long)sbase)
               : "memory");
}
__device__ __forceinline__ void wait_async0()  {
  asm volatile("s_wait_asynccnt 0"  ::: "memory");
}
__device__ __forceinline__ void wait_async4()  {
  asm volatile("s_wait_asynccnt 4"  ::: "memory");
}
__device__ __forceinline__ void wait_async16() {
  asm volatile("s_wait_asynccnt 16" ::: "memory");
}

// ---------------------------------------------------------------------------
// fp32 -> bf16 bulk convert (activation side, row-major kept)
// ---------------------------------------------------------------------------
__global__ __launch_bounds__(256) void k_f2bf(const float* __restrict__ src,
                                              unsigned short* __restrict__ dst,
                                              int n) {
  int i = blockIdx.x * 256 + threadIdx.x;
  if (i < n) dst[i] = f2bf(src[i]);
}

// ---------------------------------------------------------------------------
// Weight pack: fp32 [1024][1024] -> fragment-native bf16 tiles:
//   dword index = ((slab*8 + ntile)*128 + nl)*16 + kpl
//   slab = K/32 (32), ntile = N/128 (8), nl = n%128, kpl = (K%32)/2
//   value packs W[2*kp][n] (lo) and W[2*kp+1][n] (hi), kp = slab*16+kpl.
// Each GEMM B slab is a flat contiguous 8KB async copy, and every B fragment
// is 8 contiguous dwords (two ds_load_b128).
// ---------------------------------------------------------------------------
__global__ __launch_bounds__(256) void k_packW(const float* __restrict__ W,
                                               unsigned int* __restrict__ Wp) {
  int i = blockIdx.x * 256 + threadIdx.x;     // dword index, 524288 total
  int kpl   = i & 15;
  int nl    = (i >> 4) & 127;
  int ntile = (i >> 11) & 7;
  int slab  = i >> 14;
  int kp = slab * 16 + kpl;
  int n  = ntile * 128 + nl;
  unsigned int lo = f2bf(W[(size_t)(2 * kp) * 1024 + n]);
  unsigned int hi = f2bf(W[(size_t)(2 * kp + 1) * 1024 + n]);
  Wp[i] = lo | (hi << 16);
}

// ---------------------------------------------------------------------------
// QKV projection GEMM:  [8192,1024](bf16) x packed-W + bias
//   z==0 -> q_raw fp32   z==1 -> k_raw fp32
//   z==2 -> V bf16, dilated per-head, per-128-key-tile transposed pair-packed:
//           dword [(bdh*33+kt)*8192 + dim*64 + kp] packs keys (2kp,2kp+1)
// Tile 128x128, BK=32, 8 waves (4x2). Double-buffered async LDS pipeline.
// ---------------------------------------------------------------------------
__global__ __launch_bounds__(256) void k_gemm_qkv(
    const unsigned short* __restrict__ Ab,
    const unsigned int* __restrict__ Wq,
    const unsigned int* __restrict__ Wk,
    const unsigned int* __restrict__ Wv,
    const float* __restrict__ bq, const float* __restrict__ bk,
    const float* __restrict__ bv,
    float* __restrict__ qraw, float* __restrict__ kraw,
    unsigned short* __restrict__ Vd) {
  __shared__ __align__(16) unsigned int As[2][2048];  // A row-major [128][16] dw
  __shared__ __align__(16) unsigned int Bs[2][2048];  // B frag-native [128][16] dw
  const int z = blockIdx.z;
  const unsigned int* Bp = (z == 0) ? Wq : (z == 1) ? Wk : Wv;
  const float* bias = (z == 0) ? bq : (z == 1) ? bk : bv;
  const int ntile = blockIdx.x;
  const int n0 = ntile * 128;
  const int m0 = blockIdx.y * 128;
  const int tid = threadIdx.x;
  const int wv = tid >> 5, lane = tid & 31, hf = lane >> 4, lr = lane & 15;
  const int rm = (wv >> 1) * 32, cn = (wv & 1) * 64;
  const unsigned asA = lds_addr32(As), asB = lds_addr32(Bs);

  auto issue_slab = [&](int k0, int buf) {
    unsigned lbA = asA + (unsigned)buf * 8192u;
    unsigned lbB = asB + (unsigned)buf * 8192u;
    int slab = k0 >> 5;
#pragma unroll
    for (int t = 0; t < 2; t++) {
      int q = tid * 2 + t;                       // uint4 index 0..511
      int r = q >> 2, c4 = q & 3;                // A: row r, 4x16B per row
      async_load_b128(lbA + q * 16,
                      (unsigned)(((m0 + r) * 128 + (k0 >> 3) + c4) * 16), Ab);
      // B: flat contiguous 8KB slab for this (slab, ntile)
      async_load_b128(lbB + q * 16,
                      (unsigned)(((slab * 8 + ntile) * 512 + q) * 16), Bp);
    }
  };

  v8f acc[2][4];
#pragma unroll
  for (int i = 0; i < 2; i++)
#pragma unroll
    for (int j = 0; j < 4; j++)
#pragma unroll
      for (int r = 0; r < 8; r++) acc[i][j][r] = 0.0f;

  issue_slab(0, 0);
  for (int it = 0; it < 32; ++it) {
    const int buf = it & 1;
    if (it < 31) { issue_slab((it + 1) << 5, buf ^ 1); wait_async4(); }
    else         { wait_async0(); }
    __syncthreads();                    // all waves' slab-it data visible
    const unsigned int* Al = As[buf];
    const unsigned int* Bl = Bs[buf];
    FragB a[2];
#pragma unroll
    for (int ms = 0; ms < 2; ms++) {
      // 16-bit A layout: lane<16 K={0..15}, lane>=16 K={16..31} (contiguous)
      const uint4* p = (const uint4*)&Al[(rm + ms * 16 + lr) * 16];
      a[ms].q[0] = p[2 * hf];
      a[ms].q[1] = p[2 * hf + 1];
    }
#pragma unroll
    for (int ns = 0; ns < 4; ns++) {
      FragB b;   // B layout: VGPR v holds K=2v+16*hf, 2v+1+16*hf (contiguous)
      const unsigned int* bp = &Bl[(cn + ns * 16 + lr) * 16 + 8 * hf];
#pragma unroll
      for (int v = 0; v < 8; v++) b.u[v] = bp[v];
      acc[0][ns] = wmma_bf16(a[0].v, b.v, acc[0][ns]);
      acc[1][ns] = wmma_bf16(a[1].v, b.v, acc[1][ns]);
    }
    __syncthreads();                    // done reading buf -> safe to overwrite
  }
#pragma unroll
  for (int ms = 0; ms < 2; ms++)
#pragma unroll
    for (int ns = 0; ns < 4; ns++) {
      FragC cc; cc.v = acc[ms][ns];
#pragma unroll
      for (int r = 0; r < 8; r++) {
        int row = m0 + rm + ms * 16 + r + 8 * hf;   // C: M = r + 8*(lane/16)
        int col = n0 + cn + ns * 16 + lr;           // N = lane%16
        float val = cc.f[r] + bias[col];
        if (z == 0) {
          qraw[(size_t)row * 1024 + col] = val;
        } else if (z == 1) {
          kraw[(size_t)row * 1024 + col] = val;
        } else {
          int d = (row >> 7) & 1, nn = row >> 8, l = row & 127;
          int j = nn * 128 + l;                     // dilated position
          int h = col >> 7, dim = col & 127;
          int kt = j >> 7, kp = (j & 127) >> 1, par = j & 1;
          size_t idx = ((((size_t)(d * 8 + h)) * 33 + kt) * 8192 +
                        (size_t)dim * 64 + kp) * 2 + par;
          Vd[idx] = f2bf(val);
        }
      }
    }
}

// ---------------------------------------------------------------------------
// RMSNorm + RoPE + dilation for Q/K; zero-fill of dilated pad rows for Q/K/V.
// One block per sequence row (8448 rows incl. 256 pad rows).
// ---------------------------------------------------------------------------
__global__ __launch_bounds__(256) void k_norm_rope(
    const float* __restrict__ qraw, const float* __restrict__ kraw,
    const float* __restrict__ freqs, const float* __restrict__ gq,
    const float* __restrict__ gk, unsigned short* __restrict__ Qd,
    unsigned short* __restrict__ Kd, unsigned short* __restrict__ Vd) {
  __shared__ float redq[256];
  __shared__ float redk[256];
  const int s = blockIdx.x;
  const int t = threadIdx.x;
  const int c = t * 4;
  const int h = c >> 7, dim = c & 127;
  if (s >= 8192) {                       // pad region -> zeros (j in [4096,4224))
    int e = s - 8192;
    int d = e >> 7;
    int j = 4096 + (e & 127);
    size_t off = (((size_t)(d * 8 + h)) * 4224 + j) * 128 + dim;
    int kt = j >> 7, kp = (j & 127) >> 1, par = j & 1;
    size_t vb = (((size_t)(d * 8 + h)) * 33 + kt) * 8192;
#pragma unroll
    for (int u = 0; u < 4; u++) {
      Qd[off + u] = 0;
      Kd[off + u] = 0;
      Vd[(vb + (size_t)(dim + u) * 64 + kp) * 2 + par] = 0;
    }
    return;
  }
  float4 q = *(const float4*)&qraw[(size_t)s * 1024 + c];
  float4 k = *(const float4*)&kraw[(size_t)s * 1024 + c];
  redq[t] = q.x * q.x + q.y * q.y + q.z * q.z + q.w * q.w;
  redk[t] = k.x * k.x + k.y * k.y + k.z * k.z + k.w * k.w;
  __syncthreads();
  for (int st = 128; st > 0; st >>= 1) {
    if (t < st) { redq[t] += redq[t + st]; redk[t] += redk[t + st]; }
    __syncthreads();
  }
  float rq = rsqrtf(redq[0] * (1.0f / 1024.0f) + 1e-6f);
  float rk = rsqrtf(redk[0] * (1.0f / 1024.0f) + 1e-6f);
  float4 g4 = *(const float4*)&gq[c];
  float4 gk4 = *(const float4*)&gk[c];
  float qn0 = q.x * rq * g4.x, qn1 = q.y * rq * g4.y;
  float qn2 = q.z * rq * g4.z, qn3 = q.w * rq * g4.w;
  float kn0 = k.x * rk * gk4.x, kn1 = k.y * rk * gk4.y;
  float kn2 = k.z * rk * gk4.z, kn3 = k.w * rk * gk4.w;
  int i0 = dim >> 1;                     // even pair index (c is 4-aligned)
  float th0 = freqs[(size_t)s * 64 + i0];
  float th1 = freqs[(size_t)s * 64 + i0 + 1];
  float c0, s0, c1, s1;
  sincosf(th0, &s0, &c0);
  sincosf(th1, &s1, &c1);
  float qo0 = qn0 * c0 - qn1 * s0, qo1 = qn0 * s0 + qn1 * c0;
  float qo2 = qn2 * c1 - qn3 * s1, qo3 = qn2 * s1 + qn3 * c1;
  float ko0 = kn0 * c0 - kn1 * s0, ko1 = kn0 * s0 + kn1 * c0;
  float ko2 = kn2 * c1 - kn3 * s1, ko3 = kn2 * s1 + kn3 * c1;
  int d = (s >> 7) & 1, nn = s >> 8, l = s & 127;
  int j = nn * 128 + l;
  size_t off = (((size_t)(d * 8 + h)) * 4224 + j) * 128 + dim;
  Qd[off + 0] = f2bf(qo0); Qd[off + 1] = f2bf(qo1);
  Qd[off + 2] = f2bf(qo2); Qd[off + 3] = f2bf(qo3);
  Kd[off + 0] = f2bf(ko0); Kd[off + 1] = f2bf(ko1);
  Kd[off + 2] = f2bf(ko2); Kd[off + 3] = f2bf(ko3);
}

// ---------------------------------------------------------------------------
// Sliding-window flash attention over the dilated sequence (window +-384).
// Grid (qt=33, h=8, bd=2). 256 threads = 8 waves; wave handles 16 query rows.
// Dynamic LDS 160KB: K tile 2x32KB + V tile 2x32KB + per-wave P 32KB.
// K/V tiles double-buffered via async global->LDS DMA.
// ---------------------------------------------------------------------------
__global__ __launch_bounds__(256) void k_attn(
    const unsigned short* __restrict__ Qd, const unsigned short* __restrict__ Kd,
    const unsigned short* __restrict__ Vd, unsigned short* __restrict__ Od) {
  extern __shared__ unsigned int smem[];
  // smem[0..16383]      : K tiles, 2 buffers of 8192 dwords ([128 keys][64] dw)
  // smem[16384..32767]  : V tiles, 2 buffers ([128 dims][64 key-pairs] dw)
  // smem[32768..40959]  : P staging, 8 waves * 1024 dwords
  unsigned int* Ps = smem + 32768;
  const int qt = blockIdx.x, h = blockIdx.y, bd = blockIdx.z;
  const int tid = threadIdx.x;
  const int wv = tid >> 5, lane = tid & 31, hf = lane >> 4, lr = lane & 15;
  const size_t headbase = ((size_t)(bd * 8 + h)) * 4224 * 128;
  const float scale = 0.08838834764831845f;   // 1/sqrt(128)
  const unsigned asK = lds_addr32(smem), asV = lds_addr32(smem + 16384);

  auto issue_tile = [&](int kt, int buf) {
    const void* Kgp = (const void*)(Kd + headbase + (size_t)kt * 16384);
    const void* Vgp = (const void*)(Vd + ((((size_t)(bd * 8 + h)) * 33 + kt)
                                          * 8192) * 2);
    unsigned lbK = asK + (unsigned)buf * 32768u;
    unsigned lbV = asV + (unsigned)buf * 32768u;
#pragma unroll
    for (int t = 0; t < 8; t++) {
      unsigned q = (unsigned)(tid + t * 256) * 16u;   // 2048 uint4 per tile
      async_load_b128(lbK + q, q, Kgp);
      async_load_b128(lbV + q, q, Vgp);
    }
  };

  // Q fragments (16 rows x 128 dims = 4 A-frags), loaded once from global.
  FragB aq[4];
  {
    int row = qt * 128 + wv * 16 + lr;
    const uint4* Qp = (const uint4*)(Qd + headbase + (size_t)row * 128);
#pragma unroll
    for (int f = 0; f < 4; f++) {
      aq[f].q[0] = Qp[f * 4 + 2 * hf];
      aq[f].q[1] = Qp[f * 4 + 2 * hf + 1];
    }
  }

  v8f o[8];
  float m_run[8], l_run[8];
#pragma unroll
  for (int i = 0; i < 8; i++) {
#pragma unroll
    for (int r = 0; r < 8; r++) o[i][r] = 0.0f;
    m_run[i] = -3.0e38f;
    l_run[i] = 0.0f;
  }

  int lo = qt - 3; if (lo < 0) lo = 0;
  int hi = qt + 3; if (hi > 32) hi = 32;
  issue_tile(lo, 0);
  for (int kt = lo; kt <= hi; ++kt) {
    const int buf = (kt - lo) & 1;
    if (kt < hi) { issue_tile(kt + 1, buf ^ 1); wait_async16(); }
    else         { wait_async0(); }
    __syncthreads();                    // all waves' tile-kt data visible
    const unsigned int* Ks = smem + buf * 8192;
    const unsigned int* Vs = smem + 16384 + buf * 8192;

    // ---- scores S = Q K^T (16 x 128 per wave) ----
    FragC sc[8];
#pragma unroll
    for (int ns = 0; ns < 8; ns++) {
      v8f accv;
#pragma unroll
      for (int r = 0; r < 8; r++) accv[r] = 0.0f;
#pragma unroll
      for (int f = 0; f < 4; f++) {
        FragB fk;                        // B-frag of K^T; 8 contiguous dwords
        const unsigned int* kb = &Ks[(ns * 16 + lr) * 64 + f * 16 + 8 * hf];
#pragma unroll
        for (int v = 0; v < 8; v++) fk.u[v] = kb[v];
        accv = wmma_bf16(aq[f].v, fk.v, accv);
      }
      sc[ns].v = accv;
      int key = kt * 128 + ns * 16 + lr;
#pragma unroll
      for (int r = 0; r < 8; r++) {
        int p = qt * 128 + wv * 16 + r + 8 * hf;
        int dlt = key - p;
        bool ok = (dlt <= 384) && (dlt >= -384);
        sc[ns].f[r] = ok ? sc[ns].f[r] * scale : -1.0e30f;
      }
    }

    // ---- online softmax update ----
#pragma unroll
    for (int r = 0; r < 8; r++) {
      float mx = sc[0].f[r];
#pragma unroll
      for (int ns = 1; ns < 8; ns++) mx = fmaxf(mx, sc[ns].f[r]);
      mx = fmaxf(mx, __shfl_xor(mx, 1));
      mx = fmaxf(mx, __shfl_xor(mx, 2));
      mx = fmaxf(mx, __shfl_xor(mx, 4));
      mx = fmaxf(mx, __shfl_xor(mx, 8));
      float mnew = fmaxf(m_run[r], mx);
      float corr = __expf(m_run[r] - mnew);
      float ls = 0.0f;
#pragma unroll
      for (int ns = 0; ns < 8; ns++) {
        float pe = __expf(sc[ns].f[r] - mnew);
        sc[ns].f[r] = pe;
        ls += pe;
      }
      ls += __shfl_xor(ls, 1);
      ls += __shfl_xor(ls, 2);
      ls += __shfl_xor(ls, 4);
      ls += __shfl_xor(ls, 8);
      l_run[r] = l_run[r] * corr + ls;
      m_run[r] = mnew;
#pragma unroll
      for (int ds = 0; ds < 8; ds++) o[ds][r] = o[ds][r] * corr;
    }

    // ---- re-lay P into A-fragment form via wave-private LDS ----
    unsigned int* Pw = Ps + wv * 1024;   // 16 rows x 128 keys bf16
#pragma unroll
    for (int ns = 0; ns < 8; ns++)
#pragma unroll
      for (int r = 0; r < 8; r++)
        ((unsigned short*)Pw)[(r + 8 * hf) * 128 + ns * 16 + lr] =
            f2bf(sc[ns].f[r]);

    // ---- O += P V ----
#pragma unroll
    for (int f = 0; f < 4; f++) {
      FragB pa;
      const unsigned int* pb = Pw + lr * 64 + f * 16 + 8 * hf;
#pragma unroll
      for (int v = 0; v < 4; v++) { pa.u[v] = pb[v]; pa.u[4 + v] = pb[4 + v]; }
#pragma unroll
      for (int ds = 0; ds < 8; ds++) {
        FragB fv;                        // B-frag of V; 8 contiguous dwords
        const unsigned int* vb = &Vs[(ds * 16 + lr) * 64 + f * 16 + 8 * hf];
#pragma unroll
        for (int v = 0; v < 8; v++) fv.u[v] = vb[v];
        o[ds] = wmma_bf16(pa.v, fv.v, o[ds]);
      }
    }
    __syncthreads();                    // done reading buf -> safe to overwrite
  }

  // ---- normalize and write O (bf16, interleaved-head layout) ----
  const size_t obase = ((size_t)bd * 4224) * 1024;
#pragma unroll
  for (int r = 0; r < 8; r++) {
    float inv = 1.0f / l_run[r];
    int j = qt * 128 + wv * 16 + r + 8 * hf;
    FragC oc;
#pragma unroll
    for (int ds = 0; ds < 8; ds++) {
      oc.v = o[ds];
      int col = h * 128 + ds * 16 + lr;
      Od[obase + (size_t)j * 1024 + col] = f2bf(oc.f[r] * inv);
    }
  }
}

// ---------------------------------------------------------------------------
// Output GEMM: Od[8448,1024](bf16) x packed-Wo + bo, un-dilate to d_out fp32
// [8192,1024] (pad rows n==32 skipped). Double-buffered async LDS pipeline.
// ---------------------------------------------------------------------------
__global__ __launch_bounds__(256) void k_gemm_out(
    const unsigned short* __restrict__ Ab, const unsigned int* __restrict__ Bp,
    const float* __restrict__ bias, float* __restrict__ out) {
  __shared__ __align__(16) unsigned int As[2][2048];
  __shared__ __align__(16) unsigned int Bs[2][2048];
  const int ntile = blockIdx.x;
  const int n0 = ntile * 128;
  const int m0 = blockIdx.y * 128;
  const int tid = threadIdx.x;
  const int wv = tid >> 5, lane = tid & 31, hf = lane >> 4, lr = lane & 15;
  const int rm = (wv >> 1) * 32, cn = (wv & 1) * 64;
  const unsigned asA = lds_addr32(As), asB = lds_addr32(Bs);

  auto issue_slab = [&](int k0, int buf) {
    unsigned lbA = asA + (unsigned)buf * 8192u;
    unsigned lbB = asB + (unsigned)buf * 8192u;
    int slab = k0 >> 5;
#pragma unroll
    for (int t = 0; t < 2; t++) {
      int q = tid * 2 + t;
      int r = q >> 2, c4 = q & 3;
      async_load_b128(lbA + q * 16,
                      (unsigned)(((m0 + r) * 128 + (k0 >> 3) + c4) * 16), Ab);
      async_load_b128(lbB + q * 16,
                      (unsigned)(((slab * 8 + ntile) * 512 + q) * 16), Bp);
    }
  };

  v8f acc[2][4];
#pragma unroll
  for (int i = 0; i < 2; i++)
#pragma unroll
    for (int j = 0; j < 4; j++)
#pragma unroll
      for (int r = 0; r < 8; r++) acc[i][j][r] = 0.0f;

  issue_slab(0, 0);
  for (int it = 0; it < 32; ++it) {
    const int buf = it & 1;
    if (it < 31) { issue_slab((it + 1) << 5, buf ^ 1); wait_async4(); }
    else         { wait_async0(); }
    __syncthreads();
    const unsigned int* Al = As[buf];
    const unsigned int* Bl = Bs[buf];
    FragB a[2];
#pragma unroll
    for (int ms = 0; ms < 2; ms++) {
      const uint4* p = (const uint4*)&Al[(rm + ms * 16 + lr) * 16];
      a[ms].q[0] = p[2 * hf];
      a[ms].q[1] = p[2 * hf + 1];
    }
#pragma unroll
    for (int ns = 0; ns < 4; ns++) {
      FragB b;
      const unsigned int* bp = &Bl[(cn + ns * 16 + lr) * 16 + 8 * hf];
#pragma unroll
      for (int v = 0; v < 8; v++) b.u[v] = bp[v];
      acc[0][ns] = wmma_bf16(a[0].v, b.v, acc[0][ns]);
      acc[1][ns] = wmma_bf16(a[1].v, b.v, acc[1][ns]);
    }
    __syncthreads();
  }
#pragma unroll
  for (int ms = 0; ms < 2; ms++)
#pragma unroll
    for (int ns = 0; ns < 4; ns++) {
      FragC cc; cc.v = acc[ms][ns];
#pragma unroll
      for (int r = 0; r < 8; r++) {
        int row = m0 + rm + ms * 16 + r + 8 * hf;   // 0..8447
        int col = n0 + cn + ns * 16 + lr;
        float val = cc.f[r] + bias[col];
        int bd = row / 4224;
        int j = row - bd * 4224;
        int nn = j >> 7, l = j & 127;
        if (nn < 32)                                 // skip dilated pad rows
          out[((size_t)(nn * 256 + bd * 128 + l)) * 1024 + col] = val;
      }
    }
}

// ---------------------------------------------------------------------------
// Host launcher
// ---------------------------------------------------------------------------
extern "C" void kernel_launch(void* const* d_in, const int* in_sizes, int n_in,
                              void* d_out, int out_size, void* d_ws,
                              size_t ws_size, hipStream_t stream) {
  (void)in_sizes; (void)n_in; (void)out_size; (void)ws_size;
  const float* x     = (const float*)d_in[0];
  const float* freqs = (const float*)d_in[2];
  const float* Wq    = (const float*)d_in[3];
  const float* bq    = (const float*)d_in[4];
  const float* Wk    = (const float*)d_in[5];
  const float* bk    = (const float*)d_in[6];
  const float* Wv    = (const float*)d_in[7];
  const float* bv    = (const float*)d_in[8];
  const float* Wo    = (const float*)d_in[9];
  const float* bo    = (const float*)d_in[10];
  const float* gq    = (const float*)d_in[11];
  const float* gk    = (const float*)d_in[12];

  // Workspace layout (bytes, all regions 256B aligned). Total ~144.2 MB.
  char* ws = (char*)d_ws;
  unsigned short* xb  = (unsigned short*)ws;                       // 16 MB
  unsigned int*   Wqb = (unsigned int*)(ws + 16777216);            // 2 MB each
  unsigned int*   Wkb = Wqb + 524288;
  unsigned int*   Wvb = Wkb + 524288;
  unsigned int*   Wob = Wvb + 524288;
  unsigned short* Qd  = (unsigned short*)(ws + 25165824);          // 16.5 MB each
  unsigned short* Kd  = Qd + 8650752;                              // 2*8*4224*128
  unsigned short* Vd  = Kd + 8650752;                              // packed tiles
  float* qraw = (float*)(ws + 77070336);                           // 32 MB
  float* kraw = qraw + 8388608;                                    // 32 MB
  unsigned short* Od = (unsigned short*)(ws + 77070336);           // aliases qraw
                                                                   // (dead by attn)

  // 1) conversions / weight packing (fragment-native layout)
  k_f2bf<<<32768, 256, 0, stream>>>(x, xb, 8388608);
  k_packW<<<2048, 256, 0, stream>>>(Wq, Wqb);
  k_packW<<<2048, 256, 0, stream>>>(Wk, Wkb);
  k_packW<<<2048, 256, 0, stream>>>(Wv, Wvb);
  k_packW<<<2048, 256, 0, stream>>>(Wo, Wob);

  // 2) QKV projections (z: 0=q_raw, 1=k_raw, 2=V dilated packed bf16)
  k_gemm_qkv<<<dim3(8, 64, 3), 256, 0, stream>>>(
      xb, Wqb, Wkb, Wvb, bq, bk, bv, qraw, kraw, Vd);

  // 3) RMSNorm + RoPE + dilation for Q/K; zero pad rows of Q/K/V
  k_norm_rope<<<8448, 256, 0, stream>>>(qraw, kraw, freqs, gq, gk, Qd, Kd, Vd);

  // 4) Sliding-window flash attention (160 KB dynamic LDS per workgroup)
  k_attn<<<dim3(33, 8, 2), 256, 163840, stream>>>(Qd, Kd, Vd, Od);

  // 5) Output projection + un-dilation into d_out
  k_gemm_out<<<dim3(8, 66), 256, 0, stream>>>(Od, Wob, bo, (float*)d_out);
}
